// LocationAwareAttention_25494925869663
// MI455X (gfx1250) — matrix operand order repaired and checked by reading, not compiled, not run on hardware
//
#include <hip/hip_runtime.h>
#include <hip/hip_bf16.h>

#define B_ 32
#define S_ 2048
#define D_ 512

typedef __bf16 bf16_t;
typedef bf16_t        v16bf    __attribute__((ext_vector_type(16)));
typedef float         v8f      __attribute__((ext_vector_type(8)));
typedef unsigned short ushort8v __attribute__((ext_vector_type(8)));
typedef unsigned short ushort4v __attribute__((ext_vector_type(4)));

__device__ __forceinline__ unsigned short f32_bf16(float f) {
  unsigned int u = __float_as_uint(f);
  u += 0x7FFFu + ((u >> 16) & 1u);   // round-to-nearest-even
  return (unsigned short)(u >> 16);
}

union Frag16 {
  struct { ushort8v lo; ushort8v hi; } p;
  v16bf v;
};

// ---------------- Kernel 1: qproj[b,n] = sum_k query[b,k] * w_q[k,n] ----------------
__global__ void qproj_kernel(const float* __restrict__ query,
                             const float* __restrict__ w_q,
                             float* __restrict__ qproj) {
  __shared__ float qrow[D_];
  const int b = blockIdx.x;
  const int t = threadIdx.x;          // 256 threads
  qrow[t]       = query[b * D_ + t];
  qrow[t + 256] = query[b * D_ + t + 256];
  __syncthreads();
  for (int nn = 0; nn < 2; ++nn) {
    const int n = t + nn * 256;
    float acc = 0.f;
    for (int k = 0; k < D_; ++k) acc += qrow[k] * w_q[k * D_ + n];
    qproj[b * D_ + n] = acc;
  }
}

// ---------------- Kernel 2: wvT[n][k] = bf16(w_v[k][n]) ----------------
__global__ void wvt_kernel(const float* __restrict__ w_v,
                           unsigned short* __restrict__ wvT) {
  __shared__ float tile[32][33];
  const int k0 = blockIdx.y * 32, n0 = blockIdx.x * 32;
  const int tx = threadIdx.x, ty = threadIdx.y;   // (32, 8)
  for (int i = 0; i < 4; ++i)
    tile[ty + i * 8][tx] = w_v[(size_t)(k0 + ty + i * 8) * D_ + n0 + tx];
  __syncthreads();
  for (int i = 0; i < 4; ++i) {
    const int n = ty + i * 8;
    wvT[(size_t)(n0 + n) * D_ + k0 + tx] = f32_bf16(tile[tx][n]);
  }
}

// ---------------- Kernel 3: fused value@w_v (bf16 WMMA) + loc + tanh + score dot ----------------
__global__ __launch_bounds__(256) void energy_kernel(
    const float* __restrict__ value,
    const float* __restrict__ energy,
    const float* __restrict__ conv_w,
    const float* __restrict__ conv_b,
    const float* __restrict__ bias,
    const float* __restrict__ score_w,
    const float* __restrict__ qproj,
    const unsigned short* __restrict__ wvT,
    float* __restrict__ scores) {
  __shared__ unsigned short Atile[32][D_];   // 32 KB: 32 rows of value in bf16
  __shared__ float eL[34];                   // alignment-energy halo [s0-1 .. s0+32]
  __shared__ float sred[32];                 // per-row score accumulators

  const int b    = blockIdx.y;
  const int s0   = blockIdx.x * 32;
  const int t    = threadIdx.x;              // 256 threads = 8 waves
  const int lane = t & 31;
  const int wave = t >> 5;
  const int lr   = lane & 15;
  const bool hif = lane >= 16;

  // ---- stage A tile: 32 x 512 fp32 -> bf16 in LDS (coalesced float4 loads) ----
  const float4* vsrc = (const float4*)(value + (size_t)(b * S_ + s0) * D_);
  for (int i = 0; i < 16; ++i) {
    const int fi = t + i * 256;              // 0..4095 float4 index
    const float4 v = vsrc[fi];
    const int r  = fi >> 7;                  // / 128
    const int kc = (fi & 127) << 2;
    ushort4v u;
    u[0] = f32_bf16(v.x); u[1] = f32_bf16(v.y);
    u[2] = f32_bf16(v.z); u[3] = f32_bf16(v.w);
    *(ushort4v*)&Atile[r][kc] = u;
  }
  if (t < 34) {
    const int s = s0 - 1 + t;
    eL[t] = (s >= 0 && s < S_) ? energy[b * S_ + s] : 0.f;
  }
  if (t < 32) sred[t] = 0.f;
  __syncthreads();

  const int nwbase = wave * 64;              // this wave's 64 output columns

  v8f acc[2][4];
  for (int m = 0; m < 2; ++m)
    for (int c = 0; c < 4; ++c)
      acc[m][c] = {};

  // ---- GEMM: K = 512 in steps of 32, v_wmma_f32_16x16x32_bf16 ----
  for (int kstep = 0; kstep < D_; kstep += 32) {
    // A-fragment per ISA 16-bit A 16x32 layout:
    // lanes 0-15 (M=lr): K {0..7} then {16..23}; lanes 16-31: K {8..15} then {24..31}
    Frag16 a[2];
    const int ko = kstep + (hif ? 8 : 0);
    for (int m = 0; m < 2; ++m) {
      const int row = m * 16 + lr;
      a[m].p.lo = *(const ushort8v*)&Atile[row][ko];
      a[m].p.hi = *(const ushort8v*)&Atile[row][ko + 16];
    }
    // B-fragment matching A: lanes 0-15 hold K {0..7} then {16..23} (N=lr),
    // lanes 16-31 hold K {8..15} then {24..31} of wvT[n][k].
    const int kb = kstep + (hif ? 8 : 0);
    for (int c = 0; c < 4; ++c) {
      Frag16 bf;
      const ushort8v* bp =
          (const ushort8v*)(wvT + (size_t)(nwbase + c * 16 + lr) * D_ + kb);
      bf.p.lo = bp[0];
      bf.p.hi = bp[2];
      for (int m = 0; m < 2; ++m)
        acc[m][c] = __builtin_amdgcn_wmma_f32_16x16x32_bf16(
            false, a[m].v, false, bf.v, (short)0, acc[m][c], false, false);
    }
  }

  // ---- epilogue: + qproj + loc(conv) + bias, tanh, dot with score_w ----
  float qv[4], bb[4], c0[4], c1[4], c2[4], sw[4];
  for (int c = 0; c < 4; ++c) {
    const int d = nwbase + c * 16 + lr;      // this lane's N (column) per C layout
    qv[c] = qproj[b * D_ + d];
    bb[c] = bias[d] + conv_b[d];
    c0[c] = conv_w[d * 3 + 0];
    c1[c] = conv_w[d * 3 + 1];
    c2[c] = conv_w[d * 3 + 2];
    sw[c] = score_w[d];
  }
  for (int m = 0; m < 2; ++m) {
    for (int j = 0; j < 8; ++j) {
      // C/D layout: lanes 0-15 -> M=j, lanes 16-31 -> M=8+j
      const int rl = m * 16 + (hif ? 8 + j : j);
      const float em = eL[rl], e0v = eL[rl + 1], ep = eL[rl + 2];
      float part = 0.f;
      for (int c = 0; c < 4; ++c) {
        const float h = tanhf(acc[m][c][j] + qv[c] +
                              c0[c] * em + c1[c] * e0v + c2[c] * ep + bb[c]);
        part += h * sw[c];
      }
      // reduce over the 16 lanes that share this row (xor of low 4 lane bits)
      for (int off = 1; off < 16; off <<= 1)
        part += __shfl_xor(part, off, 32);
      if (lr == 0) atomicAdd(&sred[rl], part);
    }
  }
  __syncthreads();
  if (t < 32) scores[b * S_ + s0 + t] = sred[t];
}

// ---------------- Kernel 4: softmax over S (+ zero the context slab) ----------------
__global__ void softmax_kernel(const float* __restrict__ scores,
                               float* __restrict__ out) {
  __shared__ float red[256];
  const int b = blockIdx.x;
  const int t = threadIdx.x;                  // 256 threads, 8 elems each
  float* ctx   = out;                         // [B*D]
  float* align = out + B_ * D_;               // [B*S]
  if (b == 0)
    for (int i = t; i < B_ * D_; i += 256) ctx[i] = 0.f;

  float vals[8];
  float mx = -1e30f;
  for (int i = 0; i < 8; ++i) {
    vals[i] = scores[b * S_ + t + i * 256];
    mx = fmaxf(mx, vals[i]);
  }
  red[t] = mx; __syncthreads();
  for (int s = 128; s > 0; s >>= 1) {
    if (t < s) red[t] = fmaxf(red[t], red[t + s]);
    __syncthreads();
  }
  mx = red[0]; __syncthreads();

  float sum = 0.f;
  for (int i = 0; i < 8; ++i) { vals[i] = expf(vals[i] - mx); sum += vals[i]; }
  red[t] = sum; __syncthreads();
  for (int s = 128; s > 0; s >>= 1) {
    if (t < s) red[t] += red[t + s];
    __syncthreads();
  }
  const float inv = 1.f / red[0];
  for (int i = 0; i < 8; ++i) align[b * S_ + t + i * 256] = vals[i] * inv;
}

// ---------------- Kernel 5: context[b,d] = sum_s align[b,s] * value[b,s,d] ----------------
__global__ void context_kernel(const float* __restrict__ value,
                               const float* __restrict__ align,
                               float* __restrict__ ctx) {
  __shared__ float aL[128];
  const int b = blockIdx.y;
  const int chunk = blockIdx.x;               // 16 chunks of 128 rows
  const int t = threadIdx.x;                  // 256 threads
  if (t < 128) aL[t] = align[b * S_ + chunk * 128 + t];
  __syncthreads();
  float a0 = 0.f, a1 = 0.f;
  const float* vp = value + (size_t)(b * S_ + chunk * 128) * D_;
  for (int i = 0; i < 128; ++i) {
    const float a = aL[i];
    a0 += a * vp[(size_t)i * D_ + t];
    a1 += a * vp[(size_t)i * D_ + t + 256];
  }
  atomicAdd(&ctx[b * D_ + t], a0);
  atomicAdd(&ctx[b * D_ + t + 256], a1);
}

extern "C" void kernel_launch(void* const* d_in, const int* in_sizes, int n_in,
                              void* d_out, int out_size, void* d_ws, size_t ws_size,
                              hipStream_t stream) {
  const float* query   = (const float*)d_in[0];
  const float* value   = (const float*)d_in[1];
  const float* energy  = (const float*)d_in[2];
  const float* conv_w  = (const float*)d_in[3];
  const float* conv_b  = (const float*)d_in[4];
  const float* w_q     = (const float*)d_in[5];
  const float* w_v     = (const float*)d_in[6];
  const float* bias    = (const float*)d_in[7];
  const float* score_w = (const float*)d_in[8];
  // d_in[9] = score_b: constant shift, cancels in softmax -> ignored.

  float* out = (float*)d_out;                 // [context B*D | align B*S]

  char* ws = (char*)d_ws;                     // ~832 KB total
  float* qproj = (float*)ws;          ws += (size_t)B_ * D_ * sizeof(float);
  unsigned short* wvT = (unsigned short*)ws;
                                      ws += (size_t)D_ * D_ * sizeof(unsigned short);
  float* scores = (float*)ws;         ws += (size_t)B_ * S_ * sizeof(float);

  qproj_kernel<<<B_, 256, 0, stream>>>(query, w_q, qproj);
  wvt_kernel<<<dim3(16, 16), dim3(32, 8), 0, stream>>>(w_v, wvT);
  energy_kernel<<<dim3(S_ / 32, B_), 256, 0, stream>>>(
      value, energy, conv_w, conv_b, bias, score_w, qproj, wvT, scores);
  softmax_kernel<<<B_, 256, 0, stream>>>(scores, out);
  context_kernel<<<dim3(16, B_), 256, 0, stream>>>(value, out + B_ * D_, out);
}
